// GraphSAGEProd_18562848654083
// MI455X (gfx1250) — matrix operand order repaired and checked by reading, not compiled, not run on hardware
//
#include <hip/hip_runtime.h>
#include <hip/hip_bf16.h>

#define NN   100000
#define NE   640000
#define INDIM 256
#define HID  128
#define BN_EPS 1e-5f

typedef __attribute__((ext_vector_type(16))) __bf16 bf16x16;
typedef __attribute__((ext_vector_type(8)))  float  floatx8;

union Frag {
    bf16x16 v;
    unsigned u[8];
    uint4 q[2];
};

__device__ __forceinline__ unsigned short f32_to_bf16_bits(float f) {
    __hip_bfloat16 h = __float2bfloat16(f);
    unsigned short s;
    __builtin_memcpy(&s, &h, 2);
    return s;
}
__device__ __forceinline__ unsigned pack2bf(float lo, float hi) {
    return ((unsigned)f32_to_bf16_bits(hi) << 16) | (unsigned)f32_to_bf16_bits(lo);
}

// ---------------- utility kernels ----------------
__global__ void k_zero(float* __restrict__ p, int n) {
    int i = blockIdx.x * blockDim.x + threadIdx.x;
    if (i < n) p[i] = 0.f;
}

// Pack W (K x 128, f32 row-major) into WMMA B-fragment order:
// element t: j=t&15, lane=(t>>4)&31, frag=t>>9, nt=frag&7, kt=frag>>3
// lane holds col N = lane&15, K = kt*32 + (j>>3)*16 + (lane>>4)*8 + (j&7)  (dense 16-bit B layout)
__global__ void k_pack_w(const float* __restrict__ W, unsigned short* __restrict__ out, int K) {
    int t = blockIdx.x * blockDim.x + threadIdx.x;
    if (t >= K * HID) return;
    int j    = t & 15;
    int lane = (t >> 4) & 31;
    int frag = t >> 9;
    int nt   = frag & 7;
    int kt   = frag >> 3;
    int col  = lane & 15;
    int kh   = lane >> 4;
    int k    = kt * 32 + (j >> 3) * 16 + kh * 8 + (j & 7);
    out[t] = f32_to_bf16_bits(W[(size_t)k * HID + nt * 16 + col]);
}

__global__ void k_deg(const long long* __restrict__ ei, float* __restrict__ deg) {
    int e = blockIdx.x * blockDim.x + threadIdx.x;
    if (e < NE) atomicAdd(&deg[(int)ei[NE + e]], 1.0f);
}
__global__ void k_invdeg(float* __restrict__ deg) {
    int i = blockIdx.x * blockDim.x + threadIdx.x;
    if (i < NN) deg[i] = 1.0f / fmaxf(deg[i], 1.0f);
}

// ---------------- pre GEMM: H = relu(X @ Wpre + b), K=256 ----------------
__global__ __launch_bounds__(128) void k_pre_gemm(const float* __restrict__ X,
        const unsigned short* __restrict__ Bp, const float* __restrict__ bias,
        float* __restrict__ H) {
    __shared__ unsigned ldsA[16 * 129];   // 16 rows x 128 bf16-pairs, +1 dword pad
    const int r0  = blockIdx.x * 16;
    const int tid = threadIdx.x;
    const float2* x2 = (const float2*)X;
#pragma unroll
    for (int it = 0; it < 16; ++it) {
        int idx = tid + it * 128;             // 0..2047
        int row = idx >> 7, kp = idx & 127;
        float2 f = x2[(size_t)(r0 + row) * 128 + kp];
        ldsA[row * 129 + kp] = pack2bf(f.x, f.y);
    }
    __syncthreads();
    const int wave = tid >> 5, lane = tid & 31;
    const int row = lane & 15, kh = lane >> 4;
    floatx8 c0 = {}; floatx8 c1 = {};
#pragma unroll
    for (int kt = 0; kt < 8; ++kt) {
        const unsigned* ar = &ldsA[row * 129 + kt * 16];
        Frag a;
#pragma unroll
        for (int v = 0; v < 4; ++v) { a.u[v] = ar[kh * 4 + v]; a.u[4 + v] = ar[8 + kh * 4 + v]; }
        Frag b0, b1;
        const uint4* p0 = (const uint4*)(Bp + (((kt * 8 + wave * 2    ) * 32 + lane) << 4));
        const uint4* p1 = (const uint4*)(Bp + (((kt * 8 + wave * 2 + 1) * 32 + lane) << 4));
        b0.q[0] = p0[0]; b0.q[1] = p0[1];
        b1.q[0] = p1[0]; b1.q[1] = p1[1];
        c0 = __builtin_amdgcn_wmma_f32_16x16x32_bf16(false, a.v, false, b0.v, (short)0, c0, false, false);
        c1 = __builtin_amdgcn_wmma_f32_16x16x32_bf16(false, a.v, false, b1.v, (short)0, c1, false, false);
    }
    const int col0 = wave * 32 + (lane & 15), col1 = col0 + 16;
    const float bb0 = bias[col0], bb1 = bias[col1];
#pragma unroll
    for (int v = 0; v < 8; ++v) {
        int m = r0 + v + kh * 8;
        H[(size_t)m * HID + col0] = fmaxf(c0[v] + bb0, 0.f);
        H[(size_t)m * HID + col1] = fmaxf(c1[v] + bb1, 0.f);
    }
}

// ---------------- edge scatter: Agg[dst] += H[src], one wave per edge ----------------
__global__ __launch_bounds__(256) void k_scatter(const long long* __restrict__ ei,
        const float* __restrict__ Hin, float* __restrict__ Agg) {
    int gw   = (blockIdx.x * blockDim.x + threadIdx.x) >> 5;
    int lane = threadIdx.x & 31;
    if (gw >= NE) return;
    long long s = ei[gw], d = ei[NE + gw];
    float4 v = *(const float4*)&Hin[(size_t)s * HID + lane * 4];
    float* dp = &Agg[(size_t)d * HID + lane * 4];
    atomicAdd(dp + 0, v.x); atomicAdd(dp + 1, v.y);
    atomicAdd(dp + 2, v.z); atomicAdd(dp + 3, v.w);
}

// ---------------- layer GEMM: Hout = (Agg*invdeg)@Wl + Hin@Wr + b, + BN stats ----------------
__global__ __launch_bounds__(128) void k_layer_gemm(const float* __restrict__ Agg,
        const float* __restrict__ Hin, const float* __restrict__ invdeg,
        const unsigned short* __restrict__ Bl, const unsigned short* __restrict__ Br,
        const float* __restrict__ bias, float* __restrict__ Hout,
        float* __restrict__ colsum, float* __restrict__ colsq) {
    __shared__ unsigned ldsL[16 * 65];
    __shared__ unsigned ldsR[16 * 65];
    const int r0  = blockIdx.x * 16;
    const int tid = threadIdx.x;
    const float2* a2 = (const float2*)Agg;
    const float2* h2 = (const float2*)Hin;
#pragma unroll
    for (int it = 0; it < 8; ++it) {
        int idx = tid + it * 128;             // 0..1023
        int row = idx >> 6, kp = idx & 63;
        float s  = invdeg[r0 + row];
        float2 f = a2[(size_t)(r0 + row) * 64 + kp];
        ldsL[row * 65 + kp] = pack2bf(f.x * s, f.y * s);
        float2 g = h2[(size_t)(r0 + row) * 64 + kp];
        ldsR[row * 65 + kp] = pack2bf(g.x, g.y);
    }
    __syncthreads();
    const int wave = tid >> 5, lane = tid & 31;
    const int row = lane & 15, kh = lane >> 4;
    floatx8 c0 = {}; floatx8 c1 = {};
#pragma unroll
    for (int mtx = 0; mtx < 2; ++mtx) {
        const unsigned* lds = mtx ? ldsR : ldsL;
        const unsigned short* Bp = mtx ? Br : Bl;
#pragma unroll
        for (int kt = 0; kt < 4; ++kt) {
            const unsigned* ar = &lds[row * 65 + kt * 16];
            Frag a;
#pragma unroll
            for (int v = 0; v < 4; ++v) { a.u[v] = ar[kh * 4 + v]; a.u[4 + v] = ar[8 + kh * 4 + v]; }
            Frag b0, b1;
            const uint4* p0 = (const uint4*)(Bp + (((kt * 8 + wave * 2    ) * 32 + lane) << 4));
            const uint4* p1 = (const uint4*)(Bp + (((kt * 8 + wave * 2 + 1) * 32 + lane) << 4));
            b0.q[0] = p0[0]; b0.q[1] = p0[1];
            b1.q[0] = p1[0]; b1.q[1] = p1[1];
            c0 = __builtin_amdgcn_wmma_f32_16x16x32_bf16(false, a.v, false, b0.v, (short)0, c0, false, false);
            c1 = __builtin_amdgcn_wmma_f32_16x16x32_bf16(false, a.v, false, b1.v, (short)0, c1, false, false);
        }
    }
    const int col0 = wave * 32 + (lane & 15), col1 = col0 + 16;
    const float bb0 = bias[col0], bb1 = bias[col1];
    float s0 = 0.f, q0 = 0.f, s1 = 0.f, q1 = 0.f;
#pragma unroll
    for (int v = 0; v < 8; ++v) {
        int m = r0 + v + kh * 8;
        float t0 = c0[v] + bb0, t1 = c1[v] + bb1;
        Hout[(size_t)m * HID + col0] = t0;
        Hout[(size_t)m * HID + col1] = t1;
        s0 += t0; q0 += t0 * t0; s1 += t1; q1 += t1 * t1;
    }
    s0 += __shfl_down(s0, 16, 32); q0 += __shfl_down(q0, 16, 32);
    s1 += __shfl_down(s1, 16, 32); q1 += __shfl_down(q1, 16, 32);
    if (lane < 16) {
        atomicAdd(&colsum[col0], s0); atomicAdd(&colsq[col0], q0);
        atomicAdd(&colsum[col1], s1); atomicAdd(&colsq[col1], q1);
    }
}

__global__ void k_bn_final(const float* __restrict__ colsum, const float* __restrict__ colsq,
                           const float* __restrict__ gamma, const float* __restrict__ beta,
                           float* __restrict__ scale, float* __restrict__ shift) {
    int c = threadIdx.x;
    if (c >= HID) return;
    const float invN = 1.0f / (float)NN;
    float mean = colsum[c] * invN;
    float var  = fmaxf(colsq[c] * invN - mean * mean, 0.f);
    float sc   = gamma[c] * rsqrtf(var + BN_EPS);
    scale[c] = sc;
    shift[c] = beta[c] - mean * sc;
}

__global__ void k_bn_apply(const float* __restrict__ Hin, const float* __restrict__ scale,
                           const float* __restrict__ shift, float* __restrict__ Hout) {
    int i = blockIdx.x * blockDim.x + threadIdx.x;     // over NN*32 float4s
    if (i >= NN * 32) return;
    float4 h = ((const float4*)Hin)[i];
    int c = (i & 31) * 4;
    h.x = fmaxf(h.x * scale[c + 0] + shift[c + 0], 0.f);
    h.y = fmaxf(h.y * scale[c + 1] + shift[c + 1], 0.f);
    h.z = fmaxf(h.z * scale[c + 2] + shift[c + 2], 0.f);
    h.w = fmaxf(h.w * scale[c + 3] + shift[c + 3], 0.f);
    ((float4*)Hout)[i] = h;
}

// ---------------- head: y = H @ Whead + b, one wave per node ----------------
__global__ __launch_bounds__(256) void k_head(const float* __restrict__ H,
        const float* __restrict__ Wh, const float* __restrict__ bh, float* __restrict__ Y) {
    int gw   = (blockIdx.x * blockDim.x + threadIdx.x) >> 5;
    int lane = threadIdx.x & 31;
    if (gw >= NN) return;
    float4 h = *(const float4*)&H[(size_t)gw * HID + lane * 4];
    float4 w = *(const float4*)&Wh[lane * 4];
    float s = h.x * w.x + h.y * w.y + h.z * w.z + h.w * w.w;
    for (int off = 16; off > 0; off >>= 1) s += __shfl_down(s, off, 32);
    if (lane == 0) Y[gw] = s + bh[0];
}

extern "C" void kernel_launch(void* const* d_in, const int* in_sizes, int n_in,
                              void* d_out, int out_size, void* d_ws, size_t ws_size,
                              hipStream_t stream) {
    (void)in_sizes; (void)n_in; (void)out_size; (void)ws_size;
    const float*     x      = (const float*)d_in[0];
    const long long* ei     = (const long long*)d_in[1];
    const float*     W_pre  = (const float*)d_in[2];
    const float*     b_pre  = (const float*)d_in[3];
    const float*     W_l    = (const float*)d_in[4];
    const float*     b_l    = (const float*)d_in[5];
    const float*     W_r    = (const float*)d_in[6];
    const float*     gamma  = (const float*)d_in[7];
    const float*     beta   = (const float*)d_in[8];
    const float*     W_head = (const float*)d_in[9];
    const float*     b_head = (const float*)d_in[10];

    float* y_out = (float*)d_out;
    float* h_out = y_out + NN;           // [NN,128] final features, also layer-1 lin buffer

    char* w = (char*)d_ws;
    float* X       = (float*)w; w += (size_t)NN * HID * 4;   // h buffer (in-place per layer)
    float* Agg     = (float*)w; w += (size_t)NN * HID * 4;
    float* deg     = (float*)w; w += (size_t)NN * 4;         // becomes inv_deg
    float* colsum  = (float*)w; w += HID * 4;
    float* colsq   = (float*)w; w += HID * 4;
    float* bnscale = (float*)w; w += HID * 4;
    float* bnshift = (float*)w; w += HID * 4;
    unsigned short* packPre = (unsigned short*)w; w += (size_t)INDIM * HID * 2;
    unsigned short* packL0  = (unsigned short*)w; w += (size_t)HID * HID * 2;
    unsigned short* packL1  = (unsigned short*)w; w += (size_t)HID * HID * 2;
    unsigned short* packR0  = (unsigned short*)w; w += (size_t)HID * HID * 2;
    unsigned short* packR1  = (unsigned short*)w; w += (size_t)HID * HID * 2;

    const int ELEM = NN * HID;

    // weight packing (bf16, WMMA fragment order)
    k_pack_w<<<(INDIM * HID + 255) / 256, 256, 0, stream>>>(W_pre, packPre, INDIM);
    k_pack_w<<<(HID * HID + 255) / 256, 256, 0, stream>>>(W_l,              packL0, HID);
    k_pack_w<<<(HID * HID + 255) / 256, 256, 0, stream>>>(W_l + HID * HID,  packL1, HID);
    k_pack_w<<<(HID * HID + 255) / 256, 256, 0, stream>>>(W_r,              packR0, HID);
    k_pack_w<<<(HID * HID + 255) / 256, 256, 0, stream>>>(W_r + HID * HID,  packR1, HID);

    // degree -> inv_deg
    k_zero<<<(NN + 255) / 256, 256, 0, stream>>>(deg, NN);
    k_deg<<<(NE + 255) / 256, 256, 0, stream>>>(ei, deg);
    k_invdeg<<<(NN + 255) / 256, 256, 0, stream>>>(deg);

    // pre layer: X = relu(x @ W_pre + b_pre)
    k_pre_gemm<<<NN / 16, 128, 0, stream>>>(x, packPre, b_pre, X);

    // ---- layer 0 ----
    k_zero<<<(ELEM + 255) / 256, 256, 0, stream>>>(Agg, ELEM);
    k_zero<<<1, 256, 0, stream>>>(colsum, 2 * HID);   // colsum + colsq contiguous
    k_scatter<<<NE / 8, 256, 0, stream>>>(ei, X, Agg);
    k_layer_gemm<<<NN / 16, 128, 0, stream>>>(Agg, X, deg, packL0, packR0, b_l, X, colsum, colsq);
    k_bn_final<<<1, 128, 0, stream>>>(colsum, colsq, gamma, beta, bnscale, bnshift);
    k_bn_apply<<<(NN * 32 + 255) / 256, 256, 0, stream>>>(X, bnscale, bnshift, X);

    // ---- layer 1 ----
    k_zero<<<(ELEM + 255) / 256, 256, 0, stream>>>(Agg, ELEM);
    k_zero<<<1, 256, 0, stream>>>(colsum, 2 * HID);
    k_scatter<<<NE / 8, 256, 0, stream>>>(ei, X, Agg);
    k_layer_gemm<<<NN / 16, 128, 0, stream>>>(Agg, X, deg, packL1, packR1, b_l + HID, h_out, colsum, colsq);
    k_bn_final<<<1, 128, 0, stream>>>(colsum, colsq, gamma + HID, beta + HID, bnscale, bnshift);
    k_bn_apply<<<(NN * 32 + 255) / 256, 256, 0, stream>>>(h_out, bnscale, bnshift, h_out);

    // head
    k_head<<<(NN * 32 + 255) / 256, 256, 0, stream>>>(h_out, W_head, b_head, y_out);
}
